// AttentionBlock_30262339568349
// MI455X (gfx1250) — matrix operand order repaired and checked by reading, not compiled, not run on hardware
//
#include <hip/hip_runtime.h>
#include <hip/hip_bf16.h>

// ---------------------------------------------------------------------------
// Transformer block on MI455X (gfx1250, wave32, WMMA + Tensor Data Mover).
//   qkv = x @ Wqkv + bqkv            -> bf16 WMMA GEMM, TDM-staged LDS tiles
//   attn: causal flash attention     -> WMMA QK^T / PV, TDM-staged V tiles
//   x2  = x + attn_out
//   h   = relu(x2 @ W1 + b1)         -> bf16 WMMA GEMM (fused ReLU)
//   out = x2 + h @ W2 + b2           -> bf16 WMMA GEMM (fused residual, fp32)
// Weights are transposed+converted once so both GEMM operands are
// row-major-in-K: every LDS tile is a plain 2-D TDM copy, no transposes.
// ---------------------------------------------------------------------------

#define NEMB 1024
#define NHEAD 16
#define HDIM 64
#define SEQ 2048
#define BATCH 2
#define MTOK (BATCH * SEQ) /* 4096 */

typedef __attribute__((ext_vector_type(16))) __bf16 v16bf;
typedef __attribute__((ext_vector_type(8))) float v8f;
typedef __attribute__((ext_vector_type(16))) unsigned short v16u;
typedef __attribute__((ext_vector_type(8))) unsigned short v8u;
typedef __attribute__((ext_vector_type(4))) unsigned int u32x4;
typedef __attribute__((ext_vector_type(8))) int i32x8;
typedef __attribute__((ext_vector_type(4))) int i32x4;

__device__ __forceinline__ unsigned short f2bf(float f) {
  unsigned int u = __builtin_bit_cast(unsigned int, f);
  u += 0x7fffu + ((u >> 16) & 1u); // round-to-nearest-even
  return (unsigned short)(u >> 16);
}

__device__ __forceinline__ v16u cat16(v8u a, v8u b) {
  v16u r;
#pragma unroll
  for (int i = 0; i < 8; ++i) { r[i] = a[i]; r[8 + i] = b[i]; }
  return r;
}

// D = A(16x32 bf16) x B(32x16 bf16) + C(16x16 f32)
__device__ __forceinline__ v8f wmma_bf16(v16u a, v16u b, v8f c) {
  union U { v16u u; v16bf b; };
  U ua, ub;
  ua.u = a;
  ub.u = b;
  return __builtin_amdgcn_wmma_f32_16x16x32_bf16(false, ua.b, false, ub.b,
                                                 (short)0, c, false, false);
}

__device__ __forceinline__ float rmax16(float v) {
#pragma unroll
  for (int m = 1; m < 16; m <<= 1) v = fmaxf(v, __shfl_xor(v, m, 32));
  return v;
}
__device__ __forceinline__ float rsum16(float v) {
#pragma unroll
  for (int m = 1; m < 16; m <<= 1) v += __shfl_xor(v, m, 32);
  return v;
}

// low 32 bits of a generic LDS address == LDS byte offset (aperture scheme)
__device__ __forceinline__ unsigned lds_addr_of(const void* p) {
  return (unsigned)(size_t)p;
}

// ---------------------------------------------------------------------------
// Tensor Data Mover: 2-D tile load, 16-bit elements, LDS row padding.
// D# bit layout per CDNA5 ISA ch.8 (group0 128b, group1 256b; groups 2/3 and
// the trailing group are zero for 2-D tiles). pad_interval_code: row length
// in DWORDs = 2<<code; pad_amount_code: (code+1) DWORDs appended/interval.
// This toolchain exposes the 6-arg builtin form.
// ---------------------------------------------------------------------------
__device__ __forceinline__ void tdm_load_2d(unsigned lds_byte,
                                            const unsigned short* gptr,
                                            unsigned tile_w, unsigned tile_h,
                                            unsigned tensor_w,
                                            unsigned tensor_h,
                                            unsigned stride_elems,
                                            unsigned pad_interval_code,
                                            unsigned pad_amount_code) {
  unsigned long long ga = (unsigned long long)(const void*)gptr;
  u32x4 g0;
  g0[0] = 1u;            // count=1, user descriptor
  g0[1] = lds_byte;      // lds_addr
  g0[2] = (unsigned)ga;  // global_addr[31:0]
  g0[3] = (unsigned)((ga >> 32) & 0x01ffffffu) | (2u << 30); // addr[56:32]|type=2
  i32x8 g1;
  g1[0] = (int)((1u << 16) | (1u << 20) | (pad_interval_code << 22) |
                (pad_amount_code << 25)); // data_size=2B, pad_enable
  g1[1] = (int)((tensor_w & 0xffffu) << 16);
  g1[2] = (int)(((tensor_w >> 16) & 0xffffu) | ((tensor_h & 0xffffu) << 16));
  g1[3] = (int)(((tensor_h >> 16) & 0xffffu) | ((tile_w & 0xffffu) << 16));
  g1[4] = (int)(tile_h & 0xffffu); // tile_dim1; tile_dim2=0 (2-D)
  g1[5] = (int)stride_elems;       // tensor_dim0_stride[31:0]
  g1[6] = 0;
  g1[7] = 0;
  i32x4 z4;
  z4[0] = z4[1] = z4[2] = z4[3] = 0;
  i32x8 z8;
#pragma unroll
  for (int i = 0; i < 8; ++i) z8[i] = 0;
  __builtin_amdgcn_tensor_load_to_lds(g0, g1, z4, z4, z8, 0);
}

// ---------------------------------------------------------------------------
// fp32 -> bf16 elementwise (activations)
// ---------------------------------------------------------------------------
__global__ void cvt_f32_bf16(const float* __restrict__ s,
                             unsigned short* __restrict__ d, int n) {
  int i = blockIdx.x * blockDim.x + threadIdx.x;
  if (i < n) d[i] = f2bf(s[i]);
}

// ---------------------------------------------------------------------------
// fp32 [R][C] -> bf16 [C][R] tiled transpose+convert (weights, once)
// ---------------------------------------------------------------------------
__global__ __launch_bounds__(256) void transpose_cvt(
    const float* __restrict__ src, unsigned short* __restrict__ dst, int R,
    int C) {
  __shared__ unsigned short tile[32][33];
  const int c0 = blockIdx.x * 32, r0 = blockIdx.y * 32;
  const int tx = threadIdx.x, ty = threadIdx.y; // 32 x 8
#pragma unroll
  for (int i = 0; i < 32; i += 8)
    tile[ty + i][tx] = f2bf(src[(size_t)(r0 + ty + i) * C + c0 + tx]);
  __syncthreads();
#pragma unroll
  for (int i = 0; i < 32; i += 8)
    dst[(size_t)(c0 + ty + i) * R + r0 + tx] = tile[tx][ty + i];
}

// ---------------------------------------------------------------------------
// WMMA GEMM: C[M,N] = A[M,K] @ BT[N,K]^T + bias, fused epilogue.
// Block 128x128, BK=32, 8 waves (wave tile 64x32). Double-buffered LDS tiles
// fetched by the Tensor Data Mover (wave 0 issues, barrier releases).
// LDS rows padded to 40 halves by TDM (pad_interval=16DW, pad_amount=4DW).
// FUSE: 0 = bf16 store; 1 = ReLU + bf16; 2 = +resid, fp32 store.
// ---------------------------------------------------------------------------
template <int FUSE>
__global__ __launch_bounds__(256) void gemm_bf16_wmma(
    const unsigned short* __restrict__ A,  // [M,K] bf16
    const unsigned short* __restrict__ BT, // [N,K] bf16 (pre-transposed)
    const float* __restrict__ bias, const float* __restrict__ resid,
    unsigned short* __restrict__ outH, float* __restrict__ outF, int M, int N,
    int K) {
  __shared__ unsigned short ldsA[2][128 * 40];
  __shared__ unsigned short ldsB[2][128 * 40];
  const int t = threadIdx.x;
  const int lane = t & 31, wid = t >> 5;
  const int n = lane & 15, hi = lane >> 4;
  const int row0 = blockIdx.y * 128, col0 = blockIdx.x * 128;
  const int wr = (wid >> 2) * 64, wc = (wid & 3) * 32;
  const int koff8 = hi * 8, koff16 = hi * 16;

  v8f acc[4][2];
#pragma unroll
  for (int i = 0; i < 4; ++i)
#pragma unroll
    for (int j = 0; j < 2; ++j) acc[i][j] = 0.0f;

  const int nk = K >> 5;
  if (wid == 0) { // prologue DMA into buffer 0
    tdm_load_2d(lds_addr_of(&ldsA[0][0]), A + (size_t)row0 * K, 32, 128, K,
                M - row0, K, 3, 3);
    tdm_load_2d(lds_addr_of(&ldsB[0][0]), BT + (size_t)col0 * K, 32, 128, K,
                N - col0, K, 3, 3);
  }
  for (int ik = 0; ik < nk; ++ik) {
    const int buf = ik & 1;
    if (wid == 0) {
      if (ik + 1 < nk) { // DMA next tiles, then wait for current pair
        const int kk = (ik + 1) << 5;
        tdm_load_2d(lds_addr_of(&ldsA[buf ^ 1][0]),
                    A + (size_t)row0 * K + kk, 32, 128, K - kk, M - row0, K, 3,
                    3);
        tdm_load_2d(lds_addr_of(&ldsB[buf ^ 1][0]),
                    BT + (size_t)col0 * K + kk, 32, 128, K - kk, N - col0, K,
                    3, 3);
        __builtin_amdgcn_s_wait_tensorcnt(2);
      } else {
        __builtin_amdgcn_s_wait_tensorcnt(0);
      }
    }
    __syncthreads(); // current buffer ready for all waves

    v16u af[4], bfr[2];
#pragma unroll
    for (int i = 0; i < 4; ++i) {
      // A 16x32 fragment: lane<16 -> k {0..7,16..23}; lane>=16 -> +8
      const unsigned short* p = &ldsA[buf][(wr + i * 16 + n) * 40 + koff8];
      af[i] = cat16(*(const v8u*)p, *(const v8u*)(p + 16));
    }
#pragma unroll
    for (int j = 0; j < 2; ++j) {
      // B 32x16 fragment from BT rows: lane<16 -> k {0..7,16..23}; lane>=16 -> +8
      const unsigned short* p = &ldsB[buf][(wc + j * 16 + n) * 40 + koff8];
      bfr[j] = cat16(*(const v8u*)p, *(const v8u*)(p + 16));
    }
#pragma unroll
    for (int i = 0; i < 4; ++i)
#pragma unroll
      for (int j = 0; j < 2; ++j)
        acc[i][j] = wmma_bf16(af[i], bfr[j], acc[i][j]);
    __syncthreads(); // all waves done with this buffer before DMA reuses it
  }

  // epilogue: C layout (m = r + 8*hi, n = lane&15)
#pragma unroll
  for (int j = 0; j < 2; ++j) {
    int col = col0 + wc + j * 16 + n;
    float bv = bias[col];
#pragma unroll
    for (int i = 0; i < 4; ++i) {
#pragma unroll
      for (int r = 0; r < 8; ++r) {
        int row = row0 + wr + i * 16 + r + 8 * hi;
        size_t idx = (size_t)row * N + col;
        float v = acc[i][j][r] + bv;
        if (FUSE == 0)
          outH[idx] = f2bf(v);
        else if (FUSE == 1)
          outH[idx] = f2bf(v > 0.0f ? v : 0.0f);
        else
          outF[idx] = v + resid[idx];
      }
    }
  }
}

// ---------------------------------------------------------------------------
// Causal flash attention. qkv: bf16 [MTOK][3*NEMB]. Block = (b, head, 128 q
// rows); 8 independent waves, each owns 16 q rows; kv tiles of 32.
// V tiles are DMAed per-wave by TDM (overlaps QK^T + softmax).
// x2 = x + softmax(QK^T/8, causal) V   (fp32 out)
// ---------------------------------------------------------------------------
__global__ __launch_bounds__(256) void attn_flash_wmma(
    const unsigned short* __restrict__ qkv, const float* __restrict__ x,
    float* __restrict__ x2) {
  __shared__ unsigned short ldsV[8][32 * 72]; // per wave: [kv][d], stride 72
  __shared__ unsigned short ldsP[8][16 * 40]; // per wave: [m][kv]

  const int blk = blockIdx.x;
  const int qt = blk & 15;          // T/128 = 16 q tiles
  const int head = (blk >> 4) & 15; // 16 heads
  const int b = blk >> 8;           // batch
  const int t = threadIdx.x;
  const int lane = t & 31, wid = t >> 5;
  const int n = lane & 15, hi = lane >> 4;
  const int qb = qt * 128 + wid * 16;
  const size_t rs = 3 * NEMB; // qkv row stride (halves)
  const int koff8 = hi * 8, koff16 = hi * 16;
  const unsigned ldsV_base = lds_addr_of(&ldsV[wid][0]);

  // Q fragments (A layout, 16 rows x d=64 as two 16x32 fragments)
  const unsigned short* qp =
      qkv + (size_t)(b * SEQ + qb + n) * rs + head * HDIM;
  v16u qfrag[2];
#pragma unroll
  for (int d = 0; d < 2; ++d)
    qfrag[d] = cat16(*(const v8u*)(qp + d * 32 + koff8),
                     *(const v8u*)(qp + d * 32 + koff8 + 16));

  v8f acc[4];
#pragma unroll
  for (int tt = 0; tt < 4; ++tt) acc[tt] = 0.0f;
  float mSt[8], lSt[8];
#pragma unroll
  for (int r = 0; r < 8; ++r) {
    mSt[r] = -3.0e38f;
    lSt[r] = 0.0f;
  }

  for (int kvb = 0; kvb <= qb + 15; kvb += 32) {
    // ---- issue V tile DMA: 32 kv rows x 64 d, rows past SEQ read as 0 ----
    tdm_load_2d(ldsV_base,
                qkv + (size_t)(b * SEQ + kvb) * rs + 2 * NEMB + head * HDIM,
                64, 32, 64, (unsigned)(SEQ - kvb), 3 * NEMB, 4, 3);

    // ---- S = Q K^T for 16 q rows x 32 kv cols (two C tiles) ----
    int kr0 = kvb + n;
    if (kr0 > SEQ - 1) kr0 = SEQ - 1;
    int kr1 = kvb + 16 + n;
    if (kr1 > SEQ - 1) kr1 = SEQ - 1;
    const unsigned short* kp0 =
        qkv + (size_t)(b * SEQ + kr0) * rs + NEMB + head * HDIM + koff8;
    const unsigned short* kp1 =
        qkv + (size_t)(b * SEQ + kr1) * rs + NEMB + head * HDIM + koff8;
    v16u bk0[2], bk1[2];
#pragma unroll
    for (int d = 0; d < 2; ++d) {
      bk0[d] = cat16(*(const v8u*)(kp0 + d * 32), *(const v8u*)(kp0 + d * 32 + 16));
      bk1[d] = cat16(*(const v8u*)(kp1 + d * 32), *(const v8u*)(kp1 + d * 32 + 16));
    }
    v8f c0 = 0.0f, c1 = 0.0f;
#pragma unroll
    for (int d = 0; d < 2; ++d) {
      c0 = wmma_bf16(qfrag[d], bk0[d], c0);
      c1 = wmma_bf16(qfrag[d], bk1[d], c1);
    }

    // ---- online softmax (row-wise across the 16-lane half) ----
    float p0[8], p1[8], scl[8];
#pragma unroll
    for (int r = 0; r < 8; ++r) {
      int q = qb + r + 8 * hi;
      float s0 = c0[r] * 0.125f; // 1/sqrt(64)
      float s1 = c1[r] * 0.125f;
      if (kvb + n > q) s0 = -3.0e38f;
      if (kvb + 16 + n > q) s1 = -3.0e38f;
      float rm = rmax16(fmaxf(s0, s1));
      float mNew = fmaxf(mSt[r], rm);
      float sc = __expf(mSt[r] - mNew);
      p0[r] = __expf(s0 - mNew);
      p1[r] = __expf(s1 - mNew);
      float sum = rsum16(p0[r] + p1[r]);
      lSt[r] = lSt[r] * sc + sum;
      mSt[r] = mNew;
      scl[r] = sc;
    }

    // ---- write P (C layout) to LDS as bf16, rescale O ----
#pragma unroll
    for (int r = 0; r < 8; ++r) {
      int m = r + 8 * hi;
      ldsP[wid][m * 40 + n] = f2bf(p0[r]);
      ldsP[wid][m * 40 + 16 + n] = f2bf(p1[r]);
    }
#pragma unroll
    for (int tt = 0; tt < 4; ++tt)
#pragma unroll
      for (int r = 0; r < 8; ++r) acc[tt][r] *= scl[r];

    asm volatile("s_wait_dscnt 0" ::: "memory"); // P writes -> P reads
    __builtin_amdgcn_s_wait_tensorcnt(0);        // V tile DMA complete

    // ---- O += P(16x32) @ V(32x64) ----
    const unsigned short* pp = &ldsP[wid][n * 40 + koff16];
    v16u pfrag = cat16(*(const v8u*)pp, *(const v8u*)(pp + 8));
#pragma unroll
    for (int tt = 0; tt < 4; ++tt) {
      v16u vfrag;
#pragma unroll
      for (int i = 0; i < 16; ++i)
        vfrag[i] = ldsV[wid][(koff16 + i) * 72 + tt * 16 + n];
      acc[tt] = wmma_bf16(pfrag, vfrag, acc[tt]);
    }
  }

  // ---- epilogue: x2 = x + O / l ----
  float inv[8];
#pragma unroll
  for (int r = 0; r < 8; ++r) inv[r] = 1.0f / lSt[r];
#pragma unroll
  for (int tt = 0; tt < 4; ++tt)
#pragma unroll
    for (int r = 0; r < 8; ++r) {
      int m = r + 8 * hi;
      size_t idx =
          (size_t)(b * SEQ + qb + m) * NEMB + head * HDIM + tt * 16 + n;
      x2[idx] = x[idx] + acc[tt][r] * inv[r];
    }
}

// ---------------------------------------------------------------------------
// Host launcher
// ---------------------------------------------------------------------------
extern "C" void kernel_launch(void* const* d_in, const int* in_sizes, int n_in,
                              void* d_out, int out_size, void* d_ws,
                              size_t ws_size, hipStream_t stream) {
  const float* x = (const float*)d_in[0];    // [2,2048,1024]
  const float* Wqkv = (const float*)d_in[1]; // [1024,3072]
  const float* bqkv = (const float*)d_in[2]; // [3072]
  const float* W1 = (const float*)d_in[3];   // [1024,4096]
  const float* b1 = (const float*)d_in[4];   // [4096]
  const float* W2 = (const float*)d_in[5];   // [4096,1024]
  const float* b2 = (const float*)d_in[6];   // [1024]
  float* out = (float*)d_out;                // [2,2048,1024] fp32

  // workspace layout (total ~110 MB)
  char* w = (char*)d_ws;
  unsigned short* wqkvT = (unsigned short*)(w + 0);         //  6,291,456 B
  unsigned short* w1T = (unsigned short*)(w + 6291456);     //  8,388,608 B
  unsigned short* w2T = (unsigned short*)(w + 14680064);    //  8,388,608 B
  unsigned short* qkv_h = (unsigned short*)(w + 23068672);  // 25,165,824 B
  float* x2 = (float*)(w + 48234496);                       // 16,777,216 B
  unsigned short* h_h = (unsigned short*)(w + 65011712);    // 33,554,432 B
  unsigned short* x_h = (unsigned short*)(w + 98566144);    //  8,388,608 B
  unsigned short* x2_h = (unsigned short*)(w + 106954752);  //  8,388,608 B

  // weights: transpose + convert to bf16 [N][K]
  transpose_cvt<<<dim3(3 * NEMB / 32, NEMB / 32), dim3(32, 8), 0, stream>>>(
      Wqkv, wqkvT, NEMB, 3 * NEMB);
  transpose_cvt<<<dim3(4 * NEMB / 32, NEMB / 32), dim3(32, 8), 0, stream>>>(
      W1, w1T, NEMB, 4 * NEMB);
  transpose_cvt<<<dim3(NEMB / 32, 4 * NEMB / 32), dim3(32, 8), 0, stream>>>(
      W2, w2T, 4 * NEMB, NEMB);
  // x -> bf16
  cvt_f32_bf16<<<(MTOK * NEMB + 255) / 256, 256, 0, stream>>>(x, x_h,
                                                              MTOK * NEMB);

  // qkv = x @ Wqkv + bqkv  -> bf16
  gemm_bf16_wmma<0><<<dim3(3 * NEMB / 128, MTOK / 128), 256, 0, stream>>>(
      x_h, wqkvT, bqkv, nullptr, qkv_h, nullptr, MTOK, 3 * NEMB, NEMB);

  // x2 = x + attention(qkv)
  attn_flash_wmma<<<BATCH * NHEAD * (SEQ / 128), 256, 0, stream>>>(qkv_h, x,
                                                                   x2);
  // x2 -> bf16 for the MLP
  cvt_f32_bf16<<<(MTOK * NEMB + 255) / 256, 256, 0, stream>>>(x2, x2_h,
                                                              MTOK * NEMB);

  // h = relu(x2 @ W1 + b1) -> bf16
  gemm_bf16_wmma<1><<<dim3(4 * NEMB / 128, MTOK / 128), 256, 0, stream>>>(
      x2_h, w1T, b1, nullptr, h_h, nullptr, MTOK, 4 * NEMB, NEMB);

  // out = x2 + h @ W2 + b2 -> fp32
  gemm_bf16_wmma<2><<<dim3(NEMB / 128, MTOK / 128), 256, 0, stream>>>(
      h_h, w2T, b2, x2, nullptr, out, MTOK, NEMB, 4 * NEMB);
}
